// CLIPLoss_24421184045532
// MI455X (gfx1250) — matrix operand order repaired and checked by reading, not compiled, not run on hardware
//
#include <hip/hip_runtime.h>

typedef _Float16 f16;
typedef __attribute__((ext_vector_type(16))) _Float16 v16h;
typedef __attribute__((ext_vector_type(8)))  float    v8f;

#define NROWS 8192
#define DIM   512
#define TM    128
#define TN    128
#define TK    32
#define LDSS  40        // LDS row stride in halves (32 + 8 pad -> 20-bank stride)
#define KBLK  (NROWS / TN)   // 64 column/row blocks

union Frag16 {
    v16h h;
    uint4 q[2];
    unsigned u[8];
};

// ---------------------------------------------------------------------------
// Kernel 1: L2-normalize rows (matches F.normalize eps=1e-3) and split each
// normalized value into f16 hi + f16 lo so the WMMA GEMM reaches ~fp32 accuracy.
// grid.x = 2*NROWS (first half: image, second half: spectrum), 256 threads/row.
// ---------------------------------------------------------------------------
__global__ __launch_bounds__(256)
void norm_split(const float* __restrict__ img, const float* __restrict__ spec,
                f16* __restrict__ imgHi, f16* __restrict__ imgLo,
                f16* __restrict__ specHi, f16* __restrict__ specLo) {
    const int b   = blockIdx.x;
    const int mat = b >> 13;         // 0 = image, 1 = spectrum
    const int row = b & (NROWS - 1);
    const float* src = (mat ? spec : img) + (size_t)row * DIM;
    f16* hi = (mat ? specHi : imgHi) + (size_t)row * DIM;
    f16* lo = (mat ? specLo : imgLo) + (size_t)row * DIM;

    const int t = threadIdx.x;
    float x0 = src[t];
    float x1 = src[t + 256];

    __shared__ float red[256];
    red[t] = x0 * x0 + x1 * x1;
    __syncthreads();
    for (int off = 128; off > 0; off >>= 1) {
        if (t < off) red[t] += red[t + off];
        __syncthreads();
    }
    const float n   = sqrtf(red[0]);
    const float inv = 1.0f / fmaxf(n, 1e-3f);

    const float y0 = x0 * inv, y1 = x1 * inv;
    const f16 h0 = (f16)y0, h1 = (f16)y1;
    hi[t]       = h0;
    hi[t + 256] = h1;
    lo[t]       = (f16)(y0 - (float)h0);
    lo[t + 256] = (f16)(y1 - (float)h1);
}

// ---------------------------------------------------------------------------
// Kernel 2: fused 128x128 logits tile via v_wmma_f32_16x16x32_f16 with hi/lo
// split (3 chained WMMAs per fragment per K-step), fixed-shift exp epilogue,
// deterministic (atomic-free) row/col partial reductions, diagonal capture.
// Single LDS buffer; the fully-unrolled K-loop lets the scheduler hoist next-
// iteration global_load_b128s into the WMMA burst without spilling VGPRs.
// grid = (64, 64), block = 256 threads = 8 waves (4 M-waves x 2 N-waves).
// ---------------------------------------------------------------------------
__global__ __launch_bounds__(256)
void clip_gemm(const f16* __restrict__ imgHi, const f16* __restrict__ imgLo,
               const f16* __restrict__ specHi, const f16* __restrict__ specLo,
               const float* __restrict__ scalePtr,
               float* __restrict__ partialRow,   // [KBLK][NROWS] indexed [bn][row]
               float* __restrict__ partialCol,   // [KBLK][NROWS] indexed [bm][col]
               float* __restrict__ diag) {
    __shared__ __align__(16) f16 sA[2][TM * LDSS];   // [hi/lo][128 x 32(+pad)]
    __shared__ __align__(16) f16 sB[2][TN * LDSS];
    __shared__ float rowBuf[8][32];
    __shared__ float colBuf[8][64];

    const int tid  = threadIdx.x;
    const int lane = tid & 31;
    const int w    = tid >> 5;
    const int half = lane >> 4;      // which 16-lane half of the wave
    const int l16  = lane & 15;
    const int bn   = blockIdx.x;
    const int bm   = blockIdx.y;
    const float scale = scalePtr[0];

    const int waveM = (w & 3) * 32;  // wave's row offset inside the 128-row tile
    const int waveN = (w >> 2) * 64; // wave's col offset inside the 128-col tile

    const f16* gA[2] = { imgHi  + (size_t)bm * TM * DIM, imgLo  + (size_t)bm * TM * DIM };
    const f16* gB[2] = { specHi + (size_t)bn * TN * DIM, specLo + (size_t)bn * TN * DIM };

    v8f acc[2][4];
    const v8f vzero = { 0.f, 0.f, 0.f, 0.f, 0.f, 0.f, 0.f, 0.f };
#pragma unroll
    for (int mi = 0; mi < 2; ++mi)
#pragma unroll
        for (int ni = 0; ni < 4; ++ni) acc[mi][ni] = vzero;

#pragma unroll
    for (int kk = 0; kk < DIM; kk += TK) {
        // --- stage 128x32 hi/lo tiles of A and B into LDS (b128 transfers) ---
#pragma unroll
        for (int p = 0; p < 2; ++p) {
#pragma unroll
            for (int it = 0; it < 2; ++it) {
                const int item = tid + it * 256;       // 0..511
                const int r   = item >> 2;             // tile row 0..127
                const int c8  = (item & 3) * 8;        // half offset 0/8/16/24
                *(uint4*)&sA[p][r * LDSS + c8] =
                    *(const uint4*)(gA[p] + (size_t)r * DIM + kk + c8);
                *(uint4*)&sB[p][r * LDSS + c8] =
                    *(const uint4*)(gB[p] + (size_t)r * DIM + kk + c8);
            }
        }
        __syncthreads();

        // --- load fragments per ISA 16-bit A (16x32) / B (32x16) lane layout ---
        Frag16 aF[2][2];  // [plane][mi]
#pragma unroll
        for (int mi = 0; mi < 2; ++mi) {
            const int mrow = waveM + mi * 16 + l16;
#pragma unroll
            for (int p = 0; p < 2; ++p) {
                // VGPR r<4: K = half*8 + 2r ; VGPR r>=4: K = 16 + half*8 + 2(r-4)
                aF[p][mi].q[0] = *(const uint4*)&sA[p][mrow * LDSS + half * 8];
                aF[p][mi].q[1] = *(const uint4*)&sA[p][mrow * LDSS + 16 + half * 8];
            }
        }
        Frag16 bF[2][4];  // [plane][ni]
#pragma unroll
        for (int ni = 0; ni < 4; ++ni) {
            const int nrow = waveN + ni * 16 + l16;    // spec row = logits column
#pragma unroll
            for (int p = 0; p < 2; ++p) {
                // VGPR r<4: K = half*8 + 2r ; VGPR r>=4: K = 16 + half*8 + 2(r-4)
                bF[p][ni].q[0] = *(const uint4*)&sB[p][nrow * LDSS + half * 8];
                bF[p][ni].q[1] = *(const uint4*)&sB[p][nrow * LDSS + 16 + half * 8];
            }
        }

        // --- 3-term split product: hi*hi + hi*lo + lo*hi ---
#pragma unroll
        for (int mi = 0; mi < 2; ++mi)
#pragma unroll
            for (int ni = 0; ni < 4; ++ni) {
                v8f c = acc[mi][ni];
                c = __builtin_amdgcn_wmma_f32_16x16x32_f16(
                        false, aF[0][mi].h, false, bF[0][ni].h, (short)0, c, false, false);
                c = __builtin_amdgcn_wmma_f32_16x16x32_f16(
                        false, aF[0][mi].h, false, bF[1][ni].h, (short)0, c, false, false);
                c = __builtin_amdgcn_wmma_f32_16x16x32_f16(
                        false, aF[1][mi].h, false, bF[0][ni].h, (short)0, c, false, false);
                acc[mi][ni] = c;
            }
        __syncthreads();
    }

    // --- epilogue: fixed-shift exp + deterministic row/col reductions ---
    // C/D layout: VGPR v, lanes 0-15 -> (M=v, N=lane); lanes 16-31 -> (M=v+8, N=lane-16)
    const bool diagBlock = (bm == bn);
    float rowPart[2][8];
    float colPart[4];
#pragma unroll
    for (int mi = 0; mi < 2; ++mi)
#pragma unroll
        for (int v = 0; v < 8; ++v) rowPart[mi][v] = 0.f;
#pragma unroll
    for (int ni = 0; ni < 4; ++ni) colPart[ni] = 0.f;

#pragma unroll
    for (int mi = 0; mi < 2; ++mi) {
#pragma unroll
        for (int ni = 0; ni < 4; ++ni) {
            const v8f s = acc[mi][ni];
#pragma unroll
            for (int v = 0; v < 8; ++v) {
                const float logit = scale * s[v];
                const float e = __expf(logit - scale);   // |logit| <= scale -> safe
                // row sum over this fragment's 16 columns (within one 16-lane half)
                float rs = e;
                rs += __shfl_xor(rs, 1);
                rs += __shfl_xor(rs, 2);
                rs += __shfl_xor(rs, 4);
                rs += __shfl_xor(rs, 8);
                rowPart[mi][v] += rs;     // row = waveM + mi*16 + v + 8*half
                // column sum over this fragment's 2 rows at this v (both halves)
                const float cs = e + __shfl_xor(e, 16);
                colPart[ni] += cs;        // col = waveN + ni*16 + l16
                if (diagBlock) {
                    const int gm = waveM + mi * 16 + v + 8 * half;
                    const int gn = waveN + ni * 16 + l16;
                    if (gm == gn) diag[bm * TM + gm] = logit;
                }
            }
        }
    }

    if (l16 == 0) {       // lanes 0 and 16 each publish their half's rows
#pragma unroll
        for (int mi = 0; mi < 2; ++mi)
#pragma unroll
            for (int v = 0; v < 8; ++v)
                rowBuf[w][mi * 16 + v + 8 * half] = rowPart[mi][v];
    }
    if (half == 0) {      // lanes 0-15 publish this wave's 64 columns
#pragma unroll
        for (int ni = 0; ni < 4; ++ni)
            colBuf[w][ni * 16 + l16] = colPart[ni];
    }
    __syncthreads();

    if (tid < 128) {      // tile row r: waves r/32 and r/32+4 (fixed order sum)
        const int r = tid;
        const float rsum = rowBuf[r >> 5][r & 31] + rowBuf[(r >> 5) + 4][r & 31];
        partialRow[(size_t)bn * NROWS + bm * TM + r] = rsum;
    } else {              // tile col c: waves wc*4 + {0..3} (fixed order sum)
        const int c  = tid - 128;
        const int wc = c >> 6;
        const float csum = colBuf[wc * 4 + 0][c & 63] + colBuf[wc * 4 + 1][c & 63] +
                           colBuf[wc * 4 + 2][c & 63] + colBuf[wc * 4 + 3][c & 63];
        partialCol[(size_t)bm * NROWS + bn * TN + c] = csum;
    }
}

// ---------------------------------------------------------------------------
// Kernel 3: per-index term = 0.5*(lse_row + lse_col) - diag, fixed-order sums.
// ---------------------------------------------------------------------------
__global__ __launch_bounds__(256)
void lse_terms(const float* __restrict__ partialRow, const float* __restrict__ partialCol,
               const float* __restrict__ diag, const float* __restrict__ scalePtr,
               float* __restrict__ terms) {
    const int i = blockIdx.x * 256 + threadIdx.x;
    const float scale = scalePtr[0];
    float rs = 0.f, cs = 0.f;
    for (int b = 0; b < KBLK; ++b) {
        rs += partialRow[(size_t)b * NROWS + i];
        cs += partialCol[(size_t)b * NROWS + i];
    }
    const float lseRow = scale + logf(rs);
    const float lseCol = scale + logf(cs);
    terms[i] = 0.5f * (lseRow + lseCol) - diag[i];
}

__global__ __launch_bounds__(256)
void final_reduce(const float* __restrict__ terms, float* __restrict__ out) {
    __shared__ float red[256];
    const int t = threadIdx.x;
    float s = 0.f;
    for (int i = t; i < NROWS; i += 256) s += terms[i];
    red[t] = s;
    __syncthreads();
    for (int off = 128; off > 0; off >>= 1) {
        if (t < off) red[t] += red[t + off];
        __syncthreads();
    }
    if (t == 0) out[0] = red[0] / (float)NROWS;
}

// ---------------------------------------------------------------------------
extern "C" void kernel_launch(void* const* d_in, const int* in_sizes, int n_in,
                              void* d_out, int out_size, void* d_ws, size_t ws_size,
                              hipStream_t stream) {
    (void)in_sizes; (void)n_in; (void)out_size; (void)ws_size;
    const float* img   = (const float*)d_in[0];
    const float* spec  = (const float*)d_in[1];
    const float* scale = (const float*)d_in[2];

    const size_t FEAT = (size_t)NROWS * DIM;      // 4,194,304
    f16* imgHi  = (f16*)d_ws;
    f16* imgLo  = imgHi  + FEAT;
    f16* specHi = imgLo  + FEAT;
    f16* specLo = specHi + FEAT;
    float* partialRow = (float*)(specLo + FEAT);  // [64][8192]
    float* partialCol = partialRow + (size_t)KBLK * NROWS;
    float* diag       = partialCol + (size_t)KBLK * NROWS;
    float* terms      = diag + NROWS;

    norm_split<<<2 * NROWS, 256, 0, stream>>>(img, spec, imgHi, imgLo, specHi, specLo);

    dim3 grid(NROWS / TN, NROWS / TM);            // 64 x 64
    clip_gemm<<<grid, 256, 0, stream>>>(imgHi, imgLo, specHi, specLo, scale,
                                        partialRow, partialCol, diag);

    lse_terms<<<NROWS / 256, 256, 0, stream>>>(partialRow, partialCol, diag, scale, terms);
    final_reduce<<<1, 256, 0, stream>>>(terms, (float*)d_out);
}
